// AttentionCache_7799660610023
// MI455X (gfx1250) — matrix operand order repaired and checked by reading, not compiled, not run on hardware
//
#include <hip/hip_runtime.h>
#include <cstddef>

// ---------------- problem constants ----------------
constexpr int S_TOK     = 2048;
constexpr int D_MODEL   = 5120;
constexpr int PAST_TOK  = 1024;
constexpr int TOTAL_TOK = PAST_TOK + S_TOK;   // 3072
constexpr int H_HEADS   = 16;
constexpr int Q_LORA_D  = 1536;
constexpr int KV_LORA_D = 512;
constexpr int NOPE_D    = 128;
constexpr int ROPE_D    = 64;
constexpr int VDIM_D    = 128;
constexpr int QK_D      = NOPE_D + ROPE_D;    // 192
constexpr float EPS_C       = 1e-6f;
constexpr float SCALE_DEV_C = 0.125f;
constexpr float SM_SCALE_C  = 0.07216878364870323f;   // 1/sqrt(192)
constexpr float NEG_C       = -1.0e9f;
constexpr float LN1E4_C     = 9.210340371976184f;     // ln(10000)

// ---------------- types ----------------
typedef __bf16 bf16_t;
typedef bf16_t v16bf __attribute__((ext_vector_type(16)));
typedef bf16_t v8bf  __attribute__((ext_vector_type(8)));
typedef float  v8f   __attribute__((ext_vector_type(8)));

__device__ __forceinline__ v16bf cat16(v8bf lo, v8bf hi) {
  return __builtin_shufflevector(lo, hi, 0,1,2,3,4,5,6,7,8,9,10,11,12,13,14,15);
}
__device__ __forceinline__ v8f wmma_bf16(v16bf a, v16bf b, v8f c) {
  // v_wmma_f32_16x16x32_bf16  D = A*B + C
  return __builtin_amdgcn_wmma_f32_16x16x32_bf16(false, a, false, b, (short)0, c, false, false);
}
// A-matrix fragment (16x32 bf16): lane<16 holds K = {0..7,16..23}, lane>=16 holds K = {8..15,24..31}
__device__ __forceinline__ v16bf a_frag(const bf16_t* rowp, int lhalf) {
  const bf16_t* p = rowp + lhalf * 8;
  return cat16(*(const v8bf*)p, *(const v8bf*)(p + 16));
}
// B-matrix fragment (32x16 bf16, B = W^T): lane<16 holds K = {0..7,16..23} of col, lane>=16 holds K = {8..15,24..31}
__device__ __forceinline__ v16bf b_frag(const bf16_t* colp, int lhalf) {
  const bf16_t* p = colp + lhalf * 8;
  return cat16(*(const v8bf*)p, *(const v8bf*)(p + 16));
}
// RoPE rotation for interleaved layout: returns (even', odd->second-half')
__device__ __forceinline__ float2 rope_rot(float pos, int i, float xe, float xo) {
  float inv_freq = __expf((float)i * (-LN1E4_C / 32.0f));   // 10000^(-2i/64)
  float s, c;
  __sincosf(pos * inv_freq, &s, &c);
  return make_float2(xe * c - xo * s, xo * c + xe * s);
}

// ---------------- f32 -> bf16 convert ----------------
__global__ void f32_to_bf16_kernel(const float* __restrict__ src, bf16_t* __restrict__ dst, size_t n) {
  size_t i = (size_t)blockIdx.x * blockDim.x + threadIdx.x;
  size_t stride = (size_t)gridDim.x * blockDim.x;
  for (; i < n; i += stride) dst[i] = (bf16_t)src[i];
}

// ---------------- RMSNorm (fp32 in, bf16 out) ----------------
__global__ __launch_bounds__(256)
void rmsnorm_kernel(const float* __restrict__ x, int ldx, const float* __restrict__ w,
                    bf16_t* __restrict__ out, int ldo, int N) {
  __shared__ float red[8];
  const int row = blockIdx.x;
  const float* xr = x + (size_t)row * ldx;
  float ss = 0.f;
  for (int c = threadIdx.x; c < N; c += 256) { float v = xr[c]; ss += v * v; }
#pragma unroll
  for (int off = 16; off >= 1; off >>= 1) ss += __shfl_xor(ss, off, 32);
  if ((threadIdx.x & 31) == 0) red[threadIdx.x >> 5] = ss;
  __syncthreads();
  if (threadIdx.x < 8) {
    float v = red[threadIdx.x];
#pragma unroll
    for (int off = 4; off >= 1; off >>= 1) v += __shfl_xor(v, off, 8);
    if (threadIdx.x == 0) red[0] = v;
  }
  __syncthreads();
  const float inv = rsqrtf(red[0] / (float)N + EPS_C);
  for (int c = threadIdx.x; c < N; c += 256)
    out[(size_t)row * ldo + c] = (bf16_t)(xr[c] * inv * w[c]);
}

// ============ bf16 WMMA GEMM (narrow): C[M,N] = A[M,K] * W[N,K]^T, BN=64 (for N=576) ============
constexpr int GBM = 128, GBN = 64, GBK = 32;
constexpr int GLDA = 40, GLDB = 40;   // padded LDS row strides (bf16 elems)

__global__ __launch_bounds__(256)
void gemm_bf16_kernel(const bf16_t* __restrict__ A, const bf16_t* __restrict__ B,
                      float* __restrict__ C, int M, int N, int K,
                      const float* __restrict__ residual, float res_scale) {
  __shared__ bf16_t As[GBM * GLDA];
  __shared__ bf16_t Bs[GBN * GLDB];
  const int tid = threadIdx.x;
  const int wave = tid >> 5, lane = tid & 31;
  const int wm = wave >> 1, wn = wave & 1;         // 4x2 wave grid -> 128x64 block tile
  const int l16 = lane & 15, lhalf = lane >> 4;
  const size_t bm = (size_t)blockIdx.y * GBM;
  const size_t bn = (size_t)blockIdx.x * GBN;

  v8f c00 = 0.f, c01 = 0.f, c10 = 0.f, c11 = 0.f;

  const int ar = tid >> 1, ah = (tid & 1) * 16;    // A stage: 128 rows x 32
  const int br = tid >> 2, bh = (tid & 3) * 8;     // B stage: 64 rows x 32

  for (int k0 = 0; k0 < K; k0 += GBK) {
    {
      const bf16_t* src = A + (bm + ar) * (size_t)K + k0 + ah;
      *(int4*)&As[ar * GLDA + ah]     = *(const int4*)src;
      *(int4*)&As[ar * GLDA + ah + 8] = *(const int4*)(src + 8);
    }
    {
      const bf16_t* src = B + (bn + br) * (size_t)K + k0 + bh;
      *(int4*)&Bs[br * GLDB + bh] = *(const int4*)src;
    }
    if (k0 + GBK < K) {  // global_prefetch_b8 for next tiles
      __builtin_prefetch(A + (bm + ar) * (size_t)K + k0 + GBK + ah, 0, 1);
      __builtin_prefetch(B + (bn + br) * (size_t)K + k0 + GBK + bh, 0, 1);
    }
    __syncthreads();

    const bf16_t* a0p = &As[(wm * 32 + l16) * GLDA];
    const bf16_t* b0p = &Bs[(wn * 32 + l16) * GLDB];
    v16bf a0 = a_frag(a0p, lhalf);
    v16bf a1 = a_frag(a0p + 16 * GLDA, lhalf);
    v16bf b0 = b_frag(b0p, lhalf);
    v16bf b1 = b_frag(b0p + 16 * GLDB, lhalf);

    c00 = wmma_bf16(a0, b0, c00);
    c01 = wmma_bf16(a0, b1, c01);
    c10 = wmma_bf16(a1, b0, c10);
    c11 = wmma_bf16(a1, b1, c11);
    __syncthreads();
  }

  const size_t c0 = bn + wn * 32 + l16;
  const size_t c1 = c0 + 16;
#pragma unroll
  for (int i = 0; i < 8; ++i) {
    size_t r0 = bm + wm * 32 + i + lhalf * 8;      // C tile row map: i + 8*(lane>=16)
    size_t r1 = r0 + 16;
    float a00 = c00[i], a01 = c01[i], a10 = c10[i], a11 = c11[i];
    if (residual) {
      a00 += res_scale * residual[r0 * N + c0];
      a01 += res_scale * residual[r0 * N + c1];
      a10 += res_scale * residual[r1 * N + c0];
      a11 += res_scale * residual[r1 * N + c1];
    }
    C[r0 * N + c0] = a00;  C[r0 * N + c1] = a01;
    C[r1 * N + c0] = a10;  C[r1 * N + c1] = a11;
  }
}

// ============ bf16 WMMA GEMM (wide): 128x128 block, 32x64 wave tile, GBK=64 ============
// 16 WMMAs per barrier pair per wave; fragment reuse: 2 A-frags x 4 B-frags -> 8 WMMA / 6 loads.
constexpr int G2BM = 128, G2BN = 128, G2BK = 64;
constexpr int G2LD = 72;   // padded LDS row stride (bf16 elems): 144B -> 36-bank row spread

__global__ __launch_bounds__(256)
void gemm_bf16_128_kernel(const bf16_t* __restrict__ A, const bf16_t* __restrict__ B,
                          float* __restrict__ C, int M, int N, int K,
                          const float* __restrict__ residual, float res_scale) {
  __shared__ bf16_t As[G2BM * G2LD];
  __shared__ bf16_t Bs[G2BN * G2LD];
  const int tid = threadIdx.x;
  const int wave = tid >> 5, lane = tid & 31;
  const int wm = wave >> 1, wn = wave & 1;         // 4x2 wave grid; wave tile 32(M) x 64(N)
  const int l16 = lane & 15, lhalf = lane >> 4;
  const size_t bm = (size_t)blockIdx.y * G2BM;
  const size_t bn = (size_t)blockIdx.x * G2BN;

  v8f acc[2][4];
#pragma unroll
  for (int i = 0; i < 2; ++i)
#pragma unroll
    for (int j = 0; j < 4; ++j) acc[i][j] = 0.f;

  // stage: each thread moves 32 elems of A and 32 of B (row = tid>>1, 32-col half = tid&1)
  const int sr = tid >> 1, sh = (tid & 1) * 32;

  for (int k0 = 0; k0 < K; k0 += G2BK) {
    const bf16_t* srcA = A + (bm + sr) * (size_t)K + k0 + sh;
    const bf16_t* srcB = B + (bn + sr) * (size_t)K + k0 + sh;
#pragma unroll
    for (int e = 0; e < 4; ++e) {
      *(int4*)&As[sr * G2LD + sh + e * 8] = *(const int4*)(srcA + e * 8);
      *(int4*)&Bs[sr * G2LD + sh + e * 8] = *(const int4*)(srcB + e * 8);
    }
    if (k0 + G2BK < K) {  // global_prefetch_b8 for next stage
      __builtin_prefetch(srcA + G2BK, 0, 1);
      __builtin_prefetch(srcB + G2BK, 0, 1);
    }
    __syncthreads();

#pragma unroll
    for (int kk = 0; kk < 2; ++kk) {          // two 32-deep k-chunks per stage
      const int kc = kk * 32;
      const bf16_t* a0p = &As[(wm * 32 + l16) * G2LD + kc];
      v16bf a0 = a_frag(a0p, lhalf);
      v16bf a1 = a_frag(a0p + 16 * G2LD, lhalf);
      v16bf bf[4];
#pragma unroll
      for (int j = 0; j < 4; ++j)
        bf[j] = b_frag(&Bs[(wn * 64 + j * 16 + l16) * G2LD + kc], lhalf);
#pragma unroll
      for (int j = 0; j < 4; ++j) acc[0][j] = wmma_bf16(a0, bf[j], acc[0][j]);
#pragma unroll
      for (int j = 0; j < 4; ++j) acc[1][j] = wmma_bf16(a1, bf[j], acc[1][j]);
    }
    __syncthreads();
  }

#pragma unroll
  for (int ti = 0; ti < 2; ++ti)
#pragma unroll
    for (int j = 0; j < 4; ++j) {
      const size_t col = bn + wn * 64 + j * 16 + l16;
#pragma unroll
      for (int i = 0; i < 8; ++i) {
        const size_t row = bm + wm * 32 + ti * 16 + i + lhalf * 8;
        float v = acc[ti][j][i];
        if (residual) v += res_scale * residual[row * N + col];
        C[row * N + col] = v;
      }
    }
}

// ---------------- RoPE on q, write bf16 [H][S][QK] ----------------
__global__ void rope_q_kernel(const float* __restrict__ q, bf16_t* __restrict__ qbf) {
  const int s = blockIdx.x;
  const int c = threadIdx.x;
  if (c >= QK_D) return;
  const float pos = (float)(PAST_TOK + s);
  for (int h = 0; h < H_HEADS; ++h) {
    const float* qs = q + (size_t)s * (H_HEADS * QK_D) + h * QK_D;
    float val;
    if (c < NOPE_D) {
      val = qs[c];
    } else {
      int i = c - NOPE_D;
      if (i < 32) { float2 r = rope_rot(pos, i, qs[NOPE_D + 2 * i], qs[NOPE_D + 2 * i + 1]); val = r.x; }
      else { int j = i - 32; float2 r = rope_rot(pos, j, qs[NOPE_D + 2 * j], qs[NOPE_D + 2 * j + 1]); val = r.y; }
    }
    qbf[((size_t)h * S_TOK + s) * QK_D + c] = (bf16_t)val;
  }
}

// ---------------- assemble present_k/present_v (fp32 out) + bf16 mirrors ----------------
__global__ void build_kv_kernel(const float* __restrict__ past_k, const float* __restrict__ past_v,
                                const float* __restrict__ kv, const float* __restrict__ ckv,
                                float* __restrict__ pk, float* __restrict__ pv,
                                bf16_t* __restrict__ kbf, bf16_t* __restrict__ vbf) {
  const int t = blockIdx.x, h = blockIdx.y, c = threadIdx.x;   // 192 threads
  float kval = 0.f, vval = 0.f;
  if (t < PAST_TOK) {
    if (c < QK_D)   kval = past_k[((size_t)h * PAST_TOK + t) * QK_D + c];
    if (c < VDIM_D) vval = past_v[((size_t)h * PAST_TOK + t) * VDIM_D + c];
  } else {
    const int s = t - PAST_TOK;
    if (c < NOPE_D) {
      kval = kv[(size_t)s * (H_HEADS * (NOPE_D + VDIM_D)) + h * (NOPE_D + VDIM_D) + c];
    } else if (c < QK_D) {
      int i = c - NOPE_D;
      const float* kpe = ckv + (size_t)s * (KV_LORA_D + ROPE_D) + KV_LORA_D;
      const float pos = (float)t;
      if (i < 32) { float2 r = rope_rot(pos, i, kpe[2 * i], kpe[2 * i + 1]); kval = r.x; }
      else { int j = i - 32; float2 r = rope_rot(pos, j, kpe[2 * j], kpe[2 * j + 1]); kval = r.y; }
    }
    if (c < VDIM_D)
      vval = kv[(size_t)s * (H_HEADS * (NOPE_D + VDIM_D)) + h * (NOPE_D + VDIM_D) + NOPE_D + c];
  }
  if (c < QK_D)   { size_t idx = ((size_t)h * TOTAL_TOK + t) * QK_D + c;   pk[idx] = kval; kbf[idx] = (bf16_t)kval; }
  if (c < VDIM_D) { size_t idx = ((size_t)h * TOTAL_TOK + t) * VDIM_D + c; pv[idx] = vval; vbf[idx] = (bf16_t)vval; }
}

// ---------------- flash attention: 128 queries/block (8 waves x 16), 64-key tiles ----------------
constexpr int ABM = 128, ABN = 64;
constexpr int LDK = 200, LDV = 72, LDP = 72;   // padded LDS strides

__global__ __launch_bounds__(256)
void attention_kernel(const bf16_t* __restrict__ qbf, const bf16_t* __restrict__ kbf,
                      const bf16_t* __restrict__ vbf, bf16_t* __restrict__ obf) {
  __shared__ bf16_t Ks[ABN * LDK];            // 64 x 192 K tile (row-major)
  __shared__ bf16_t Vts[VDIM_D * LDV];        // V tile transposed: [n][k]
  __shared__ bf16_t Ps[8][16 * LDP];          // per-wave P tile (bf16, row-major)

  const int h   = blockIdx.y;
  const int qb0 = blockIdx.x * ABM;
  const int tid = threadIdx.x;
  const int wave = tid >> 5, lane = tid & 31;
  const int l16 = lane & 15, lhalf = lane >> 4;
  const int rowoff = lhalf * 8;               // WMMA C row offset for this lane half

  // Q fragments resident in registers: 16 rows x 192 -> 6 A-frags
  v16bf qa[6];
  {
    const bf16_t* qrow = qbf + ((size_t)h * S_TOK + qb0 + wave * 16 + l16) * QK_D;
#pragma unroll
    for (int kk = 0; kk < 6; ++kk) qa[kk] = a_frag(qrow + kk * 32, lhalf);
  }

  v8f o[8];
#pragma unroll
  for (int j = 0; j < 8; ++j) o[j] = 0.f;
  float m_i[8], l_i[8];
#pragma unroll
  for (int i = 0; i < 8; ++i) { m_i[i] = -3.0e38f; l_i[i] = 0.f; }

  const int qpos_base = PAST_TOK + qb0 + wave * 16;
  const int nt = (PAST_TOK + qb0 + ABM) / ABN;   // causal: keys up to PAST+qb0+127

  for (int kt = 0; kt < nt; ++kt) {
    const int t0 = kt * ABN;
    // stage K tile (64x192): 1536 8-elem chunks / 256 threads
#pragma unroll
    for (int j = 0; j < 6; ++j) {
      int cidx = j * 256 + tid;
      int r = cidx / 24, c8 = (cidx % 24) * 8;
      *(v8bf*)&Ks[r * LDK + c8] =
          *(const v8bf*)&kbf[((size_t)h * TOTAL_TOK + t0 + r) * QK_D + c8];
    }
    // stage V tile transposed (64x128 -> [128][64]): 1024 chunks / 256 threads
#pragma unroll
    for (int j = 0; j < 4; ++j) {
      int cidx = j * 256 + tid;
      int r = cidx >> 4, n8 = (cidx & 15) * 8;
      v8bf vv = *(const v8bf*)&vbf[((size_t)h * TOTAL_TOK + t0 + r) * VDIM_D + n8];
#pragma unroll
      for (int e = 0; e < 8; ++e) Vts[(n8 + e) * LDV + r] = vv[e];
    }
    __syncthreads();

    // scores S[16 x 64] = Q * K^T : 4 WMMA column tiles, K-dim chained over 6 frags
    v8f sc[4];
#pragma unroll
    for (int nb = 0; nb < 4; ++nb) {
      v8f s = 0.f;
#pragma unroll
      for (int kk = 0; kk < 6; ++kk) {
        v16bf b = b_frag(&Ks[(nb * 16 + l16) * LDK + kk * 32], lhalf);
        s = wmma_bf16(qa[kk], b, s);
      }
      sc[nb] = s;
    }
    // scale + causal mask (in registers; C-tile layout: row = i+rowoff, col = nb*16+l16)
#pragma unroll
    for (int nb = 0; nb < 4; ++nb) {
      const int kpos = t0 + nb * 16 + l16;
#pragma unroll
      for (int i = 0; i < 8; ++i) {
        const int qpos = qpos_base + i + rowoff;
        float v = sc[nb][i] * SM_SCALE_C;
        sc[nb][i] = (kpos <= qpos) ? v : (v + NEG_C);
      }
    }
    // online softmax: per-row max/sum via width-16 shuffles (rows live in 16-lane groups)
    float alpha[8];
#pragma unroll
    for (int i = 0; i < 8; ++i) {
      float tm = fmaxf(fmaxf(sc[0][i], sc[1][i]), fmaxf(sc[2][i], sc[3][i]));
#pragma unroll
      for (int off = 8; off >= 1; off >>= 1) tm = fmaxf(tm, __shfl_xor(tm, off, 16));
      const float mn = fmaxf(m_i[i], tm);
      alpha[i] = __expf(m_i[i] - mn);
      float rs = 0.f;
#pragma unroll
      for (int nb = 0; nb < 4; ++nb) {
        float p = __expf(sc[nb][i] - mn);
        sc[nb][i] = p;
        rs += p;
      }
#pragma unroll
      for (int off = 8; off >= 1; off >>= 1) rs += __shfl_xor(rs, off, 16);
      l_i[i] = alpha[i] * l_i[i] + rs;
      m_i[i] = mn;
    }
    // write P (bf16) row-major to per-wave LDS for A-frag reload
#pragma unroll
    for (int nb = 0; nb < 4; ++nb)
#pragma unroll
      for (int i = 0; i < 8; ++i)
        Ps[wave][(i + rowoff) * LDP + nb * 16 + l16] = (bf16_t)sc[nb][i];
    // rescale running accumulator
#pragma unroll
    for (int j = 0; j < 8; ++j)
#pragma unroll
      for (int i = 0; i < 8; ++i) o[j][i] *= alpha[i];
    // O += P[16x64] * V[64x128]
#pragma unroll
    for (int kk = 0; kk < 2; ++kk) {
      v16bf pa = a_frag(&Ps[wave][l16 * LDP + kk * 32], lhalf);
#pragma unroll
      for (int j = 0; j < 8; ++j) {
        v16bf vb = b_frag(&Vts[(j * 16 + l16) * LDV + kk * 32], lhalf);
        o[j] = wmma_bf16(pa, vb, o[j]);
      }
    }
    __syncthreads();
  }

  // epilogue: normalize by l and write bf16 [S][H*VDIM]
#pragma unroll
  for (int i = 0; i < 8; ++i) l_i[i] = 1.0f / l_i[i];
#pragma unroll
  for (int j = 0; j < 8; ++j)
#pragma unroll
    for (int i = 0; i < 8; ++i) {
      const size_t srow = qb0 + wave * 16 + i + rowoff;
      const int col = h * VDIM_D + j * 16 + l16;
      obf[srow * (H_HEADS * VDIM_D) + col] = (bf16_t)(o[j][i] * l_i[i]);
    }
}

// ---------------- host-side orchestration ----------------
extern "C" void kernel_launch(void* const* d_in, const int* in_sizes, int n_in,
                              void* d_out, int out_size, void* d_ws, size_t ws_size,
                              hipStream_t stream) {
  (void)in_sizes; (void)n_in; (void)out_size; (void)ws_size;
  const float* hidden    = (const float*)d_in[0];
  // d_in[1]=position_ids, d_in[2]=attention_mask: causal mask/positions computed analytically
  const float* past_k    = (const float*)d_in[3];
  const float* past_v    = (const float*)d_in[4];
  const float* ln_w      = (const float*)d_in[5];
  const float* q_a_w     = (const float*)d_in[6];
  const float* q_a_ln_w  = (const float*)d_in[7];
  const float* q_b_w     = (const float*)d_in[8];
  const float* kv_a_w    = (const float*)d_in[9];
  const float* kv_a_ln_w = (const float*)d_in[10];
  const float* kv_b_w    = (const float*)d_in[11];
  const float* o_w       = (const float*)d_in[12];

  float* hs = (float*)d_out;
  float* pk = hs + (size_t)S_TOK * D_MODEL;
  float* pv = pk + (size_t)H_HEADS * TOTAL_TOK * QK_D;

  // workspace layout (bump; BIG holds kv fp32 then is reused for q fp32;
  // QA holds q_a fp32 then is reused as bf16 attention output)
  char* ws = (char*)d_ws;
  size_t off = 0;
  auto alloc = [&](size_t bytes) { size_t o = off; off += (bytes + 255) & ~size_t(255); return ws + o; };
  bf16_t* WPOOL = (bf16_t*)alloc((size_t)D_MODEL * H_HEADS * VDIM_D * 2);        // max weight (o_w)
  bf16_t* Hbf   = (bf16_t*)alloc((size_t)S_TOK * D_MODEL * 2);
  float*  BIG   = (float*) alloc((size_t)S_TOK * 4096 * 4);                      // kv fp32 / q fp32
  float*  CKV   = (float*) alloc((size_t)S_TOK * (KV_LORA_D + ROPE_D) * 4);
  bf16_t* CKVN  = (bf16_t*)alloc((size_t)S_TOK * KV_LORA_D * 2);
  float*  QA    = (float*) alloc((size_t)S_TOK * Q_LORA_D * 4);                  // also attn-out bf16
  bf16_t* QAN   = (bf16_t*)alloc((size_t)S_TOK * Q_LORA_D * 2);
  bf16_t* Qbf   = (bf16_t*)alloc((size_t)H_HEADS * S_TOK * QK_D * 2);
  bf16_t* Kbf   = (bf16_t*)alloc((size_t)H_HEADS * TOTAL_TOK * QK_D * 2);
  bf16_t* Vbf   = (bf16_t*)alloc((size_t)H_HEADS * TOTAL_TOK * VDIM_D * 2);
  bf16_t* Obf   = (bf16_t*)QA;                                                   // alias (free after q_a rmsnorm)
  float*  Qmat  = BIG;                                                           // alias (free after build_kv)

  // 1) h = rmsnorm(hidden) -> bf16
  rmsnorm_kernel<<<S_TOK, 256, 0, stream>>>(hidden, D_MODEL, ln_w, Hbf, D_MODEL, D_MODEL);

  // 2) kv chain: ckv = h @ kv_a_w^T ; rmsnorm ; kv = @ kv_b_w^T
  f32_to_bf16_kernel<<<4096, 256, 0, stream>>>(kv_a_w, WPOOL, (size_t)(KV_LORA_D + ROPE_D) * D_MODEL);
  gemm_bf16_kernel<<<dim3((KV_LORA_D + ROPE_D) / GBN, S_TOK / GBM), 256, 0, stream>>>(
      Hbf, WPOOL, CKV, S_TOK, KV_LORA_D + ROPE_D, D_MODEL, nullptr, 0.f);        // N=576 -> narrow kernel
  rmsnorm_kernel<<<S_TOK, 256, 0, stream>>>(CKV, KV_LORA_D + ROPE_D, kv_a_ln_w, CKVN, KV_LORA_D, KV_LORA_D);
  f32_to_bf16_kernel<<<4096, 256, 0, stream>>>(kv_b_w, WPOOL, (size_t)H_HEADS * (NOPE_D + VDIM_D) * KV_LORA_D);
  gemm_bf16_128_kernel<<<dim3(H_HEADS * (NOPE_D + VDIM_D) / G2BN, S_TOK / G2BM), 256, 0, stream>>>(
      CKVN, WPOOL, BIG, S_TOK, H_HEADS * (NOPE_D + VDIM_D), KV_LORA_D, nullptr, 0.f);

  // 3) present_k / present_v (fp32 outputs) + bf16 mirrors, RoPE on k_pe
  build_kv_kernel<<<dim3(TOTAL_TOK, H_HEADS), 192, 0, stream>>>(
      past_k, past_v, BIG, CKV, pk, pv, Kbf, Vbf);

  // 4) q chain: q_a = h @ q_a_w^T ; rmsnorm ; q = @ q_b_w^T ; RoPE -> bf16
  f32_to_bf16_kernel<<<4096, 256, 0, stream>>>(q_a_w, WPOOL, (size_t)Q_LORA_D * D_MODEL);
  gemm_bf16_128_kernel<<<dim3(Q_LORA_D / G2BN, S_TOK / G2BM), 256, 0, stream>>>(
      Hbf, WPOOL, QA, S_TOK, Q_LORA_D, D_MODEL, nullptr, 0.f);
  rmsnorm_kernel<<<S_TOK, 256, 0, stream>>>(QA, Q_LORA_D, q_a_ln_w, QAN, Q_LORA_D, Q_LORA_D);
  f32_to_bf16_kernel<<<4096, 256, 0, stream>>>(q_b_w, WPOOL, (size_t)H_HEADS * QK_D * Q_LORA_D);
  gemm_bf16_128_kernel<<<dim3(H_HEADS * QK_D / G2BN, S_TOK / G2BM), 256, 0, stream>>>(
      QAN, WPOOL, Qmat, S_TOK, H_HEADS * QK_D, Q_LORA_D, nullptr, 0.f);
  rope_q_kernel<<<S_TOK, 256, 0, stream>>>(Qmat, Qbf);

  // 5) flash attention -> bf16 [S][H*VDIM]
  attention_kernel<<<dim3(S_TOK / ABM, H_HEADS), 256, 0, stream>>>(Qbf, Kbf, Vbf, Obf);

  // 6) hs = attn_out @ o_w^T + hidden * (1/8)
  f32_to_bf16_kernel<<<4096, 256, 0, stream>>>(o_w, WPOOL, (size_t)D_MODEL * H_HEADS * VDIM_D);
  gemm_bf16_128_kernel<<<dim3(D_MODEL / G2BN, S_TOK / G2BM), 256, 0, stream>>>(
      Obf, WPOOL, hs, S_TOK, D_MODEL, H_HEADS * VDIM_D, hidden, SCALE_DEV_C);
}
